// Mamba2Mixer_46780783788655
// MI455X (gfx1250) — matrix operand order repaired and checked by reading, not compiled, not run on hardware
//
#include <hip/hip_runtime.h>
#include <hip/hip_bf16.h>

typedef __bf16 bf16;
typedef __attribute__((ext_vector_type(16))) __bf16 v16bf;
typedef __attribute__((ext_vector_type(8)))  float  v8f;
typedef __attribute__((ext_vector_type(4)))  unsigned int v4u;
typedef __attribute__((ext_vector_type(8)))  int v8i;
typedef __attribute__((ext_vector_type(4)))  int v4i;

#define HIDD   2048
#define INTERD 4096
#define NHD    64
#define PDIM   64
#define NDIM   128
#define CSZ    256
#define NCHUNK 8
#define BSZ    2
#define SEQL   2048
#define ROWS   (BSZ*SEQL)      // 4096
#define PROJD  8512            // 2*INTER + 2*N + NH
#define CONVD  4352            // INTER + 2*N

union FragBF { v16bf v; uint4 q[2]; };

static __device__ inline v8f wmma_bf16(v16bf a, v16bf b, v8f c) {
  return __builtin_amdgcn_wmma_f32_16x16x32_bf16(false, a, false, b, (short)0, c, false, false);
}

// A-fragment (16x32 bf16): lane = row (lane&15); lanes<16 hold K 0..15, lanes>=16 hold K 16..31
static __device__ inline v16bf ld_afrag(const bf16* rowptr, int hi) {
  FragBF f;
  f.q[0] = *(const uint4*)(rowptr + hi * 16);
  f.q[1] = *(const uint4*)(rowptr + hi * 16 + 8);
  return f.v;
}
// B-fragment (32x16 bf16): lane = col (lane&15); lanes<16 hold K 0..15, lanes>=16 hold K 16..31
static __device__ inline v16bf ld_bfrag(const bf16* rowptr, int hi) {
  FragBF f;
  const bf16* p = rowptr + hi * 16;
  f.q[0] = *(const uint4*)(p);
  f.q[1] = *(const uint4*)(p + 8);
  return f.v;
}

static __device__ inline float sigmoidf_(float x) { return 1.0f / (1.0f + __expf(-x)); }

// ---------------------------------------------------------------- Tensor Data Mover 2D tile load
// Loads tile_rows x tile_k bf16 tile (row stride = row_stride_elems in memory) into LDS at
// lds_addr, inserting 16B of LDS padding after every 64B row (-> LDS row stride 80B = 40 elems).
static __device__ inline void tdm_load_tile(unsigned lds_addr, const void* gptr,
                                            unsigned tile_k, unsigned tile_rows,
                                            unsigned row_stride_elems) {
  unsigned long long ga = (unsigned long long)(uintptr_t)gptr;
  v4u g0;
  g0[0] = 1u;                                            // count=1, user descriptor
  g0[1] = lds_addr;                                      // LDS byte address
  g0[2] = (unsigned)(ga & 0xFFFFFFFFu);                  // global addr [31:0]
  g0[3] = (unsigned)((ga >> 32) & 0x01FFFFFFu) | (2u << 30); // addr[56:32] | type=2
  unsigned dim0 = row_stride_elems;                      // tensor dim0 (elements)
  unsigned dim1 = 1u << 20;                              // plenty of rows, no OOB hit
  v8i g1;
  g1[0] = (int)((1u << 16)                               // data_size = 2 bytes
              | (1u << 20)                               // pad_enable
              | (3u << 22)                               // pad_interval: 16 DWORDs (64B)
              | (3u << 25));                             // pad_amount: 4 DWORDs (16B)
  g1[1] = (int)((dim0 & 0xFFFFu) << 16);                 // tensor_dim0 low16
  g1[2] = (int)((dim0 >> 16) | ((dim1 & 0xFFFFu) << 16));// dim0 hi16 | dim1 low16
  g1[3] = (int)((dim1 >> 16) | (tile_k << 16));          // dim1 hi16 | tile_dim0
  g1[4] = (int)(tile_rows & 0xFFFFu);                    // tile_dim1 | tile_dim2=0 (2D)
  g1[5] = (int)row_stride_elems;                         // tensor_dim0_stride low32
  g1[6] = 0;                                             // stride hi16 | dim1_stride low16
  g1[7] = 0;
  v4i g2 = {1, 1, 0, 0};                                 // tensor_dim2=1, tensor_dim3=1
  v4i g3 = {0, (int)(1u << 16), 0, 0};                   // tensor_dim4=1, tile_dim4=0
#if defined(__clang_major__) && (__clang_major__ >= 23)
  v8i g4 = {0, 0, 0, 0, 0, 0, 0, 0};
  __builtin_amdgcn_tensor_load_to_lds(g0, g1, g2, g3, g4, 0);
#else
  __builtin_amdgcn_tensor_load_to_lds(g0, g1, g2, g3, 0);
#endif
}

// ---------------------------------------------------------------- converts
__global__ __launch_bounds__(256) void cvt_bf16_kernel(const float* __restrict__ src,
                                                       bf16* __restrict__ dst, int n) {
  int i = blockIdx.x * 256 + threadIdx.x;
  if (i < n) dst[i] = (bf16)src[i];
}

// src is K x N row-major f32 -> dst is N x K row-major bf16 (transposed)
__global__ __launch_bounds__(256) void transpose_bf16_kernel(const float* __restrict__ src,
                                                             bf16* __restrict__ dst,
                                                             int K, int N) {
  size_t i = (size_t)blockIdx.x * 256 + threadIdx.x;
  if (i >= (size_t)K * N) return;
  int k = (int)(i / N);
  int n = (int)(i % N);
  dst[(size_t)n * K + k] = (bf16)src[i];
}

// ---------------------------------------------------------------- GEMM (A: MxK bf16, BT: NxK bf16) -> C f32 MxN
// block tile 128(M) x 64(N), K-step 32, 8 waves in 4x2, wave tile 32x32.
// Double-buffered LDS staged by the Tensor Data Mover (async, TENSORcnt).
__global__ __launch_bounds__(256) void gemm_bf16_kernel(const bf16* __restrict__ A,
                                                        const bf16* __restrict__ BT,
                                                        float* __restrict__ C,
                                                        int M, int N, int K) {
  __shared__ bf16 sA[2][128][40];
  __shared__ bf16 sB[2][64][40];
  int nbn = N >> 6;
  int bm = blockIdx.x / nbn, bn = blockIdx.x % nbn;
  int t = threadIdx.x;
  int lane = t & 31, w = t >> 5, hi = (lane >> 4) & 1, l15 = lane & 15;
  v8f acc[2][2];
  #pragma unroll
  for (int i = 0; i < 2; i++)
    #pragma unroll
    for (int j = 0; j < 2; j++)
      acc[i][j] = (v8f){0.f, 0.f, 0.f, 0.f, 0.f, 0.f, 0.f, 0.f};

  const bf16* Ab = A + (size_t)bm * 128 * K;
  const bf16* Bb = BT + (size_t)bn * 64 * K;
  int mbase = (w >> 1) * 32, nbase = (w & 1) * 32;

  if (w == 0) {
    tdm_load_tile((unsigned)(uintptr_t)&sA[0][0][0], Ab, 32u, 128u, (unsigned)K);
    tdm_load_tile((unsigned)(uintptr_t)&sB[0][0][0], Bb, 32u, 64u, (unsigned)K);
  }

  int nsteps = K >> 5;
  int cur = 0;
  for (int i = 0; i < nsteps; i++) {
    if (w == 0) __builtin_amdgcn_s_wait_tensorcnt(0);  // current buffer landed in LDS
    __syncthreads();                                    // publish to all waves
    if (w == 0 && (i + 1) < nsteps) {                   // prefetch next tiles (overlaps compute)
      tdm_load_tile((unsigned)(uintptr_t)&sA[cur ^ 1][0][0], Ab + (size_t)(i + 1) * 32,
                    32u, 128u, (unsigned)K);
      tdm_load_tile((unsigned)(uintptr_t)&sB[cur ^ 1][0][0], Bb + (size_t)(i + 1) * 32,
                    32u, 64u, (unsigned)K);
    }
    v16bf af0 = ld_afrag(&sA[cur][mbase + l15][0], hi);
    v16bf af1 = ld_afrag(&sA[cur][mbase + 16 + l15][0], hi);
    v16bf bf0 = ld_bfrag(&sB[cur][nbase + l15][0], hi);
    v16bf bf1 = ld_bfrag(&sB[cur][nbase + 16 + l15][0], hi);
    acc[0][0] = wmma_bf16(af0, bf0, acc[0][0]);
    acc[0][1] = wmma_bf16(af0, bf1, acc[0][1]);
    acc[1][0] = wmma_bf16(af1, bf0, acc[1][0]);
    acc[1][1] = wmma_bf16(af1, bf1, acc[1][1]);
    cur ^= 1;
  }
  #pragma unroll
  for (int i = 0; i < 2; i++)
    #pragma unroll
    for (int j = 0; j < 2; j++)
      #pragma unroll
      for (int k = 0; k < 8; k++) {
        int row = bm * 128 + mbase + i * 16 + k + hi * 8;
        int col = bn * 64 + nbase + j * 16 + l15;
        C[(size_t)row * N + col] = acc[i][j][k];
      }
}

// ---------------------------------------------------------------- dt softplus + per-chunk cumsum of A*dt
__global__ __launch_bounds__(256) void dt_acum_kernel(const float* __restrict__ proj,
                                                      const float* __restrict__ dt_bias,
                                                      const float* __restrict__ A_log,
                                                      float* __restrict__ dtB,
                                                      float* __restrict__ Acum) {
  int bh = blockIdx.x;
  int b = bh >> 6, h = bh & 63;
  int t = threadIdx.x;
  __shared__ float sbuf[CSZ];
  float Ah = -__expf(A_log[h]);
  float bias = dt_bias[h];
  for (int c = 0; c < NCHUNK; ++c) {
    int l = c * CSZ + t;
    float raw = proj[(size_t)(b * SEQL + l) * PROJD + (INTERD + CONVD) + h] + bias;
    float dtv = (raw > 20.f) ? raw : __logf(1.f + __expf(raw));
    dtB[(size_t)(b * SEQL + l) * NHD + h] = dtv;
    sbuf[t] = Ah * dtv;
    __syncthreads();
    for (int off = 1; off < CSZ; off <<= 1) {
      float add = (t >= off) ? sbuf[t - off] : 0.f;
      __syncthreads();
      sbuf[t] += add;
      __syncthreads();
    }
    Acum[(((size_t)(b * NHD + h)) * NCHUNK + c) * CSZ + t] = sbuf[t];
    __syncthreads();
  }
}

// ---------------------------------------------------------------- depthwise conv + SiLU + split
__global__ __launch_bounds__(256) void conv_silu_kernel(const float* __restrict__ proj,
                                                        const float* __restrict__ convw,
                                                        const float* __restrict__ convb,
                                                        const float* __restrict__ dtB,
                                                        float* __restrict__ xf,
                                                        bf16* __restrict__ xdt,
                                                        bf16* __restrict__ Bb,
                                                        bf16* __restrict__ Cb) {
  size_t i = (size_t)blockIdx.x * 256 + threadIdx.x;  // over ROWS*CONVD
  int ch = (int)(i % CONVD);
  size_t row = i / CONVD;
  int b = (int)(row / SEQL), l = (int)(row % SEQL);
  float acc = convb[ch];
  #pragma unroll
  for (int k = 0; k < 4; k++) {
    int ls = l - 3 + k;
    if (ls >= 0) acc += convw[ch * 4 + k] * proj[((size_t)b * SEQL + ls) * PROJD + INTERD + ch];
  }
  float s = acc * sigmoidf_(acc);
  if (ch < INTERD) {
    xf[row * INTERD + ch] = s;
    xdt[row * INTERD + ch] = (bf16)(s * dtB[row * NHD + (ch >> 6)]);
  } else if (ch < INTERD + NDIM) {
    Bb[row * NDIM + (ch - INTERD)] = (bf16)s;
  } else {
    Cb[row * NDIM + (ch - INTERD - NDIM)] = (bf16)s;
  }
}

// ---------------------------------------------------------------- chunk states: state[p,n] = sum_l w[l]*xdt[l,p]*B[l,n]
__global__ __launch_bounds__(256) void states_kernel(const bf16* __restrict__ xdt,
                                                     const bf16* __restrict__ Bb,
                                                     const float* __restrict__ Acum,
                                                     float* __restrict__ states) {
  int id = blockIdx.x;          // b*(NCHUNK*NH) + c*NH + h
  int h = id & 63, c = (id >> 6) & 7, b = id >> 9;
  __shared__ float wl[CSZ];
  __shared__ bf16 sX[64][72];    // [p][l-chunk]
  __shared__ bf16 sBT[128][72];  // [n][l-chunk]
  int t = threadIdx.x, lane = t & 31, w = t >> 5, hi = (lane >> 4) & 1, l15 = lane & 15;
  const float* Ac = Acum + (((size_t)(b * NHD + h)) * NCHUNK + c) * CSZ;
  float Alast = Ac[CSZ - 1];
  wl[t] = __expf(Alast - Ac[t]);
  size_t rowBase = (size_t)b * SEQL + c * CSZ;
  v8f acc[4];
  #pragma unroll
  for (int j = 0; j < 4; j++) acc[j] = (v8f){0.f, 0.f, 0.f, 0.f, 0.f, 0.f, 0.f, 0.f};
  int ptile = w & 3;
  int ngrp = (w >> 2) * 4;
  int li = t >> 2;
  int pbase = (t & 3) * 16;
  int nbase2 = (t & 3) * 32;
  __syncthreads();
  for (int l0 = 0; l0 < CSZ; l0 += 64) {
    __syncthreads();
    {
      const bf16* xr = xdt + (rowBase + l0 + li) * INTERD + h * PDIM;
      float wv = wl[l0 + li];
      #pragma unroll
      for (int j = 0; j < 16; j++)
        sX[pbase + j][li] = (bf16)((float)xr[pbase + j] * wv);
      const bf16* br = Bb + (rowBase + l0 + li) * NDIM;
      #pragma unroll
      for (int j = 0; j < 32; j++)
        sBT[nbase2 + j][li] = br[nbase2 + j];
    }
    __syncthreads();
    #pragma unroll
    for (int ks = 0; ks < 64; ks += 32) {
      v16bf af = ld_afrag(&sX[ptile * 16 + l15][ks], hi);
      #pragma unroll
      for (int j = 0; j < 4; j++) {
        v16bf bfg = ld_bfrag(&sBT[(ngrp + j) * 16 + l15][ks], hi);
        acc[j] = wmma_bf16(af, bfg, acc[j]);
      }
    }
  }
  float* st = states + ((size_t)(b * NCHUNK + c) * NHD + h) * (PDIM * NDIM);
  #pragma unroll
  for (int j = 0; j < 4; j++)
    #pragma unroll
    for (int k = 0; k < 8; k++) {
      int p = ptile * 16 + k + hi * 8;
      int n = (ngrp + j) * 16 + l15;
      st[p * NDIM + n] = acc[j][k];
    }
}

// ---------------------------------------------------------------- inter-chunk scan
__global__ __launch_bounds__(256) void scan_kernel(const float* __restrict__ states,
                                                   const float* __restrict__ Acum,
                                                   bf16* __restrict__ prev) {
  int bh = blockIdx.x;
  int b = bh >> 6, h = bh & 63;
  int t = threadIdx.x;
  float P[32];
  #pragma unroll
  for (int j = 0; j < 32; j++) P[j] = 0.f;
  for (int c = 0; c < NCHUNK; c++) {
    size_t base = ((size_t)(b * NCHUNK + c) * NHD + h) * (PDIM * NDIM);
    #pragma unroll
    for (int j = 0; j < 32; j++) prev[base + t * 32 + j] = (bf16)P[j];
    float decay = __expf(Acum[(((size_t)(b * NHD + h)) * NCHUNK + c) * CSZ + (CSZ - 1)]);
    #pragma unroll
    for (int j = 0; j < 32; j++) P[j] = decay * P[j] + states[base + t * 32 + j];
  }
}

// ---------------------------------------------------------------- Y = Ldec*(C B^T) xdt + exp(Acum)*(C state^T) + D*x
__global__ __launch_bounds__(256) void ydiag_kernel(const bf16* __restrict__ Cb,
                                                    const bf16* __restrict__ Bb,
                                                    const bf16* __restrict__ xdt,
                                                    const bf16* __restrict__ prev,
                                                    const float* __restrict__ Acum,
                                                    const float* __restrict__ xf,
                                                    const float* __restrict__ Dp,
                                                    float* __restrict__ y) {
  int id = blockIdx.x;  // b*1024 + c*128 + half*64 + h
  int h = id & 63, half = (id >> 6) & 1, c = (id >> 7) & 7, b = id >> 10;
  __shared__ float sAcum[CSZ];
  __shared__ bf16 sXT[64][264];    // [p][s]
  __shared__ bf16 sG[8][16][40];   // per-wave relayout scratch
  int t = threadIdx.x, lane = t & 31, w = t >> 5, hi = (lane >> 4) & 1, l15 = lane & 15;
  size_t rowBase = (size_t)b * SEQL + c * CSZ;
  const float* Ac = Acum + (((size_t)(b * NHD + h)) * NCHUNK + c) * CSZ;
  sAcum[t] = Ac[t];
  {
    const bf16* xrow = xdt + (rowBase + t) * INTERD + h * PDIM;
    #pragma unroll 8
    for (int p = 0; p < 64; p++) sXT[p][t] = xrow[p];
  }
  __syncthreads();

  int r0 = half * 128 + w * 16;
  v16bf cf[4];
  const bf16* crow = Cb + (rowBase + r0 + l15) * NDIM;
  #pragma unroll
  for (int ks = 0; ks < 4; ks++) cf[ks] = ld_afrag(crow + ks * 32, hi);

  float Ar[8], eAr[8];
  #pragma unroll
  for (int k = 0; k < 8; k++) {
    Ar[k] = sAcum[r0 + k + hi * 8];
    eAr[k] = __expf(Ar[k]);
  }

  v8f acc[4];
  const bf16* pv = prev + ((size_t)(b * NCHUNK + c) * NHD + h) * (PDIM * NDIM);
  #pragma unroll
  for (int pt = 0; pt < 4; pt++) {
    acc[pt] = (v8f){0.f, 0.f, 0.f, 0.f, 0.f, 0.f, 0.f, 0.f};
    #pragma unroll
    for (int ks = 0; ks < 4; ks++) {
      v16bf bfrag = ld_bfrag(pv + (size_t)(pt * 16 + l15) * NDIM + ks * 32, hi);
      acc[pt] = wmma_bf16(cf[ks], bfrag, acc[pt]);
    }
    #pragma unroll
    for (int k = 0; k < 8; k++) acc[pt][k] *= eAr[k];
  }

  int sMax = r0 + 15;
  for (int s0 = 0; s0 <= sMax; s0 += 32) {
    #pragma unroll
    for (int sub = 0; sub < 2; sub++) {
      v8f g = (v8f){0.f, 0.f, 0.f, 0.f, 0.f, 0.f, 0.f, 0.f};
      int sb = s0 + sub * 16;
      const bf16* brow = Bb + (rowBase + sb + l15) * NDIM;
      #pragma unroll
      for (int ks = 0; ks < 4; ks++) {
        v16bf bfrag = ld_bfrag(brow + ks * 32, hi);
        g = wmma_bf16(cf[ks], bfrag, g);
      }
      int sidx = sb + l15;
      float As = sAcum[sidx];
      #pragma unroll
      for (int k = 0; k < 8; k++) {
        int r = r0 + k + hi * 8;
        float fct = (sidx <= r) ? __expf(Ar[k] - As) : 0.f;
        sG[w][k + hi * 8][sub * 16 + l15] = (bf16)(g[k] * fct);
      }
    }
    v16bf gf = ld_afrag(&sG[w][l15][0], hi);  // wave-private; LDS ops are in-order per wave
    #pragma unroll
    for (int pt = 0; pt < 4; pt++) {
      v16bf bfrag = ld_bfrag(&sXT[pt * 16 + l15][s0], hi);
      acc[pt] = wmma_bf16(gf, bfrag, acc[pt]);
    }
  }

  float Dh = Dp[h];
  #pragma unroll
  for (int pt = 0; pt < 4; pt++)
    #pragma unroll
    for (int k = 0; k < 8; k++) {
      int r = r0 + k + hi * 8;
      size_t idx = (rowBase + r) * INTERD + h * PDIM + pt * 16 + l15;
      y[idx] = acc[pt][k] + Dh * xf[idx];
    }
}

// ---------------------------------------------------------------- RMSNorm + gated SiLU
__global__ __launch_bounds__(256) void norm_gate_kernel(const float* __restrict__ y,
                                                        const float* __restrict__ proj,
                                                        const float* __restrict__ nw,
                                                        bf16* __restrict__ yg) {
  int row = blockIdx.x;
  int t = threadIdx.x;
  __shared__ float red[256];
  const float* yr = y + (size_t)row * INTERD;
  float s = 0.f;
  #pragma unroll
  for (int j = 0; j < 16; j++) {
    float v = yr[t * 16 + j];
    s += v * v;
  }
  red[t] = s;
  __syncthreads();
  for (int off = 128; off > 0; off >>= 1) {
    if (t < off) red[t] += red[t + off];
    __syncthreads();
  }
  float rstd = rsqrtf(red[0] * (1.0f / INTERD) + 1e-5f);
  const float* pr = proj + (size_t)row * PROJD;
  #pragma unroll
  for (int j = 0; j < 16; j++) {
    int i = t * 16 + j;
    float g = pr[i];
    float v = yr[i] * rstd * nw[i];
    yg[(size_t)row * INTERD + i] = (bf16)(v * g * sigmoidf_(g));
  }
}

// ---------------------------------------------------------------- launch
extern "C" void kernel_launch(void* const* d_in, const int* in_sizes, int n_in,
                              void* d_out, int out_size, void* d_ws, size_t ws_size,
                              hipStream_t stream) {
  const float* hs    = (const float*)d_in[0];
  const float* w1    = (const float*)d_in[1];
  const float* convw = (const float*)d_in[2];
  const float* convb = (const float*)d_in[3];
  const float* dtb   = (const float*)d_in[4];
  const float* alog  = (const float*)d_in[5];
  const float* Dp    = (const float*)d_in[6];
  const float* nw    = (const float*)d_in[7];
  const float* w2    = (const float*)d_in[8];
  float* out = (float*)d_out;

  char* ws = (char*)d_ws;
  size_t o = 0;
  auto take = [&](size_t bytes) { size_t r = o; o += (bytes + 255) & ~(size_t)255; return r; };
  bf16*  hsb    = (bf16*)(ws + take((size_t)ROWS * HIDD * 2));
  bf16*  w1T    = (bf16*)(ws + take((size_t)PROJD * HIDD * 2));
  bf16*  w2T    = (bf16*)(ws + take((size_t)HIDD * INTERD * 2));
  float* proj   = (float*)(ws + take((size_t)ROWS * PROJD * 4));
  float* dtB    = (float*)(ws + take((size_t)ROWS * NHD * 4));
  float* Acum   = (float*)(ws + take((size_t)BSZ * NHD * NCHUNK * CSZ * 4));
  float* xf     = (float*)(ws + take((size_t)ROWS * INTERD * 4));
  bf16*  xdt    = (bf16*)(ws + take((size_t)ROWS * INTERD * 2));
  bf16*  Bb     = (bf16*)(ws + take((size_t)ROWS * NDIM * 2));
  bf16*  Cb     = (bf16*)(ws + take((size_t)ROWS * NDIM * 2));
  float* states = (float*)(ws + take((size_t)BSZ * NCHUNK * NHD * PDIM * NDIM * 4));
  bf16*  prev   = (bf16*)(ws + take((size_t)BSZ * NCHUNK * NHD * PDIM * NDIM * 2));
  float* y      = (float*)(ws + take((size_t)ROWS * INTERD * 4));
  bf16*  yg     = (bf16*)(ws + take((size_t)ROWS * INTERD * 2));

  cvt_bf16_kernel<<<(ROWS * HIDD) / 256, 256, 0, stream>>>(hs, hsb, ROWS * HIDD);
  transpose_bf16_kernel<<<(HIDD * PROJD) / 256, 256, 0, stream>>>(w1, w1T, HIDD, PROJD);
  transpose_bf16_kernel<<<(INTERD * HIDD) / 256, 256, 0, stream>>>(w2, w2T, INTERD, HIDD);

  gemm_bf16_kernel<<<(ROWS / 128) * (PROJD / 64), 256, 0, stream>>>(hsb, w1T, proj, ROWS, PROJD, HIDD);

  dt_acum_kernel<<<BSZ * NHD, 256, 0, stream>>>(proj, dtb, alog, dtB, Acum);
  conv_silu_kernel<<<(ROWS * CONVD) / 256, 256, 0, stream>>>(proj, convw, convb, dtB, xf, xdt, Bb, Cb);

  states_kernel<<<BSZ * NCHUNK * NHD, 256, 0, stream>>>(xdt, Bb, Acum, states);
  scan_kernel<<<BSZ * NHD, 256, 0, stream>>>(states, Acum, prev);
  ydiag_kernel<<<BSZ * NCHUNK * 2 * NHD, 256, 0, stream>>>(Cb, Bb, xdt, prev, Acum, xf, Dp, y);

  norm_gate_kernel<<<ROWS, 256, 0, stream>>>(y, proj, nw, yg);
  gemm_bf16_kernel<<<(ROWS / 128) * (HIDD / 64), 256, 0, stream>>>(yg, w2T, out, ROWS, HIDD, INTERD);
}
